// SimpleGPT_70239895158932
// MI455X (gfx1250) — matrix operand order repaired and checked by reading, not compiled, not run on hardware
//
#include <hip/hip_runtime.h>
#include <math.h>

// ------------------------------------------------------------------
// Types for CDNA5 WMMA (gfx1250, wave32)
// ------------------------------------------------------------------
typedef __attribute__((ext_vector_type(16))) __bf16 v16bf;
typedef __attribute__((ext_vector_type(8)))  float  v8f;

union Frag16 { v16bf v; unsigned int u[8]; };

#define GPT_L  8
#define GPT_H  12
#define GPT_C  768
#define GPT_DH 64
#define GPT_FF 3072
#define GPT_V  32000
#define GPT_T  1024
#define GPT_B  2
#define GPT_M  (GPT_B * GPT_T)   // 2048 rows

__device__ __forceinline__ unsigned short f2bf(float x) {
  // round-to-nearest-even fp32 -> bf16
  unsigned int u = __float_as_uint(x);
  u += 0x7FFFu + ((u >> 16) & 1u);
  return (unsigned short)(u >> 16);
}

// CDNA5 async DMA: global -> LDS, 16 bytes per lane, tracked by ASYNCcnt.
__device__ __forceinline__ void async_b128(unsigned lds_off, const void* gaddr) {
  asm volatile("global_load_async_to_lds_b128 %0, %1, off"
               :: "v"(lds_off), "v"((unsigned long long)(size_t)gaddr)
               : "memory");
}
__device__ __forceinline__ void wait_async0() {
  asm volatile("s_wait_asynccnt 0x0" ::: "memory");
}
__device__ __forceinline__ unsigned lds_addr(const void* p) {
  // generic pointers to LDS carry the LDS byte offset in the low 32 bits
  return (unsigned)(size_t)p;
}

// ------------------------------------------------------------------
// Embedding: x[b,t,:] = tok_emb[idx[b,t],:] + pos_emb[t,:]   (fp32 out)
// ------------------------------------------------------------------
__global__ void embed_kernel(const int* __restrict__ idx,
                             const float* __restrict__ tok,
                             const float* __restrict__ pos,
                             float* __restrict__ x) {
  int row   = blockIdx.x;          // b*T + t
  int t     = row % GPT_T;
  int token = idx[row];
  const float* tp = tok + (long long)token * GPT_C;
  const float* pp = pos + (long long)t * GPT_C;
  float* xr = x + (long long)row * GPT_C;
  for (int c = threadIdx.x; c < GPT_C; c += blockDim.x)
    xr[c] = tp[c] + pp[c];
}

// ------------------------------------------------------------------
// LayerNorm (C=768): fp32 in, bf16 out; one block per row.
// ------------------------------------------------------------------
__global__ __launch_bounds__(256)
void ln_kernel(const float* __restrict__ x, unsigned short* __restrict__ y,
               const float* __restrict__ sc, const float* __restrict__ bi) {
  int row = blockIdx.x;
  const float* xr = x + (long long)row * GPT_C;
  unsigned short* yr = y + (long long)row * GPT_C;

  float sum = 0.f, sq = 0.f;
  for (int c = threadIdx.x; c < GPT_C; c += blockDim.x) {
    float v = xr[c];
    sum += v;
    sq  += v * v;
  }
  for (int off = 16; off >= 1; off >>= 1) {
    sum += __shfl_xor(sum, off, 32);
    sq  += __shfl_xor(sq,  off, 32);
  }
  __shared__ float ssum[8], ssq[8], stat[2];
  int wv = threadIdx.x >> 5, ln = threadIdx.x & 31;
  if (ln == 0) { ssum[wv] = sum; ssq[wv] = sq; }
  __syncthreads();
  if (threadIdx.x < 8) {
    float a = ssum[threadIdx.x], b = ssq[threadIdx.x];
    for (int off = 4; off >= 1; off >>= 1) {
      a += __shfl_xor(a, off, 32);
      b += __shfl_xor(b, off, 32);
    }
    if (threadIdx.x == 0) {
      float mean = a * (1.0f / GPT_C);
      float var  = b * (1.0f / GPT_C) - mean * mean;
      stat[0] = mean;
      stat[1] = rsqrtf(var + 1e-5f);
    }
  }
  __syncthreads();
  float mean = stat[0], inv = stat[1];
  for (int c = threadIdx.x; c < GPT_C; c += blockDim.x)
    yr[c] = f2bf((xr[c] - mean) * inv * sc[c] + bi[c]);
}

// ------------------------------------------------------------------
// Generic WMMA GEMM:  C(m,n) = sum_k A(m,k)*B(k,n)  (+bias,+residual,relu)
//  A: bf16 row-major [M,K], lda   -> async-DMA'd into LDS, double buffered
//  B: fp32, addr = gk*sb_k + (gn/dhB)*sb_h + (gn%dhB)*sb_d
//     -> prefetched to registers, converted to bf16, double buffered in LDS
//  C: addr = (m/mrowC)*sc_b + (m%mrowC)*sc_t + (n/dhC)*sc_h + (n%dhC)*sc_d
//     out_bf16 ? bf16 store : fp32 store
// Block 256 threads (8 waves), tile 128x128, k-step 32, wave tile 64x32.
// Requires M%128==0, N%128==0, K%32==0; tile bases divide mrowC/dhC (true here).
// ------------------------------------------------------------------
#define AS_STRIDE 40                       // ushorts per row (80B, 16B multiple)
#define AS_BYTES  (128 * AS_STRIDE * 2)    // one A buffer in bytes

__global__ __launch_bounds__(256)
void gemm_bf16(const unsigned short* __restrict__ A, const float* __restrict__ B,
               void* __restrict__ Cp,
               int M, int N, int K, int lda,
               long long sb_k, long long sb_h, long long sb_d, int dhB,
               int mrowC, long long sc_b, long long sc_t,
               int dhC, long long sc_h, long long sc_d,
               const float* __restrict__ bias,
               const float* __restrict__ add, int add_ld, int relu, int out_bf16) {
  __shared__ __align__(16) unsigned short As[2][128][AS_STRIDE];
  __shared__ unsigned short Bs[2][128][36];

  int tid  = threadIdx.x;
  int wv   = tid >> 5;
  int lane = tid & 31;
  int mw   = wv >> 2;        // 0..1 : 64-row stripe
  int nw   = wv & 3;         // 0..3 : 32-col stripe
  int hf   = lane >> 4;
  int lr   = lane & 15;

  int bm = blockIdx.y * 128;
  int bn = blockIdx.x * 128;

  unsigned asBase = lds_addr(&As[0][0][0]);

  // per-thread staging coordinates (k-invariant)
  int cB  = tid & 31;        // k within tile (B staging)
  int nB0 = tid >> 5;        // first n handled by this thread (B staging)
  int rA  = tid >> 1;        // unused; A staging row is j >> 2, 4 x 16B chunks per row
  long long bColOff[16];
#pragma unroll
  for (int ii = 0; ii < 16; ii++) {
    int gn = bn + nB0 + 8 * ii;
    bColOff[ii] = (long long)(gn / dhB) * sb_h + (long long)(gn % dhB) * sb_d;
  }
  (void)rA;

  v8f acc[4][2] = {};
  float breg[16];
  int ntiles = K >> 5;

  // ---- prologue: tile 0 ----
#pragma unroll
  for (int it = 0; it < 2; it++) {
    int j = tid + it * 256;
    int r = j >> 2, c4 = j & 3;
    async_b128(asBase + r * 80 + c4 * 16,
               A + (long long)(bm + r) * lda + c4 * 8);
  }
  {
    const float* Bk = B + (long long)cB * sb_k;
#pragma unroll
    for (int ii = 0; ii < 16; ii++) breg[ii] = Bk[bColOff[ii]];
#pragma unroll
    for (int ii = 0; ii < 16; ii++) Bs[0][nB0 + 8 * ii][cB] = f2bf(breg[ii]);
  }
  wait_async0();
  __syncthreads();

  // ---- software-pipelined main loop ----
  for (int t = 0; t < ntiles; t++) {
    int p = t & 1;
    int more = (t + 1 < ntiles);
    int k0n = (t + 1) << 5;

    if (more) {
      // async DMA next A tile into idle buffer (overlaps with WMMAs below)
#pragma unroll
      for (int it = 0; it < 2; it++) {
        int j = tid + it * 256;
        int r = j >> 2, c4 = j & 3;
        async_b128(asBase + (1 - p) * AS_BYTES + r * 80 + c4 * 16,
                   A + (long long)(bm + r) * lda + k0n + c4 * 8);
      }
      // launch next B tile loads into registers (latency hidden by WMMAs)
      const float* Bk = B + (long long)(k0n + cB) * sb_k;
#pragma unroll
      for (int ii = 0; ii < 16; ii++) breg[ii] = Bk[bColOff[ii]];
    }

    // ---- compute on buffer p ----
    v16bf af[4], bfg[2];
#pragma unroll
    for (int i = 0; i < 4; i++) {
      Frag16 u;
      int m = mw * 64 + i * 16 + lr;
#pragma unroll
      for (int vvi = 0; vvi < 8; vvi++) {
        int kb = (vvi < 4 ? 2 * vvi : 16 + 2 * (vvi - 4)) + hf * 8;
        u.u[vvi] = *(const unsigned int*)&As[p][m][kb];
      }
      af[i] = u.v;
    }
#pragma unroll
    for (int j = 0; j < 2; j++) {
      Frag16 u;
      int n = nw * 32 + j * 16 + lr;
#pragma unroll
      for (int vvi = 0; vvi < 8; vvi++) {
        int kb = (vvi < 4 ? 2 * vvi : 16 + 2 * (vvi - 4)) + hf * 8;
        u.u[vvi] = *(const unsigned int*)&Bs[p][n][kb];
      }
      bfg[j] = u.v;
    }
#pragma unroll
    for (int i = 0; i < 4; i++)
#pragma unroll
      for (int j = 0; j < 2; j++)
        acc[i][j] = __builtin_amdgcn_wmma_f32_16x16x32_bf16(
            false, af[i], false, bfg[j], (short)0, acc[i][j], false, false);

    if (more) {
      // convert + park next B tile; wait for next A DMA
#pragma unroll
      for (int ii = 0; ii < 16; ii++)
        Bs[1 - p][nB0 + 8 * ii][cB] = f2bf(breg[ii]);
      wait_async0();
    }
    __syncthreads();
  }

  // ---- epilogue: hoisted index mapping ----
  int nbase0 = bn + nw * 32;
  long long colT[2];
  float biasV[2] = {0.f, 0.f};
#pragma unroll
  for (int j = 0; j < 2; j++) {
    int nb = nbase0 + j * 16;
    colT[j] = (long long)(nb / dhC) * sc_h + (long long)(nb % dhC + lr) * sc_d;
    if (bias) biasV[j] = bias[nb + lr];
  }
  int mb0 = bm + mw * 64;
  long long rowQ = (long long)(mb0 / mrowC) * sc_b;
  int mr0 = mb0 % mrowC;

  float* Cf = (float*)Cp;
  unsigned short* Ch = (unsigned short*)Cp;
#pragma unroll
  for (int i = 0; i < 4; i++) {
    long long ra = rowQ + (long long)(mr0 + i * 16 + hf * 8) * sc_t;
#pragma unroll
    for (int s = 0; s < 8; s++) {
      int m = mb0 + i * 16 + s + hf * 8;
#pragma unroll
      for (int j = 0; j < 2; j++) {
        float val = acc[i][j][s] + biasV[j];
        if (add) val += add[(long long)m * add_ld + (nbase0 + j * 16 + lr)];
        if (relu) val = fmaxf(val, 0.f);
        long long ca = ra + colT[j];
        if (out_bf16) Ch[ca] = f2bf(val);
        else          Cf[ca] = val;
      }
      ra += sc_t;
    }
  }
}

// ------------------------------------------------------------------
// Flash attention (causal), q/k/v: bf16 [B,H,T,64], o: bf16 [B,T,H*64].
// Block = 128 threads (4 waves) handles one (b,h) x 64 query rows.
// Q/K tiles async-DMA'd to LDS; V transposed through LDS; WMMA both matmuls.
// ------------------------------------------------------------------
__global__ __launch_bounds__(128)
void attn_kernel(const unsigned short* __restrict__ q,
                 const unsigned short* __restrict__ kg,
                 const unsigned short* __restrict__ vg,
                 unsigned short* __restrict__ o, float scale) {
  __shared__ __align__(16) unsigned short Qs[64][72];  // [qrow][d], stride 144B
  __shared__ __align__(16) unsigned short Ks[32][72];  // [key][d]
  __shared__ unsigned short Vs[64][36];                // [d][key] (transposed)
  __shared__ unsigned short Ps[4][16][36];             // per-wave P tile

  int tid  = threadIdx.x;
  int wv   = tid >> 5;
  int lane = tid & 31;
  int hf   = lane >> 4;
  int lr   = lane & 15;

  int bh = blockIdx.y;
  int b  = bh / GPT_H;
  int h  = bh % GPT_H;
  int qbase = blockIdx.x * 64;

  const unsigned short* qp = q  + (long long)(b * GPT_H + h) * GPT_T * GPT_DH;
  const unsigned short* kp = kg + (long long)(b * GPT_H + h) * GPT_T * GPT_DH;
  const unsigned short* vp = vg + (long long)(b * GPT_H + h) * GPT_T * GPT_DH;

  unsigned qsBase = lds_addr(&Qs[0][0]);
  unsigned ksBase = lds_addr(&Ks[0][0]);

  // stage Q tile 64x64 bf16 once: 512 x b128 chunks, 4 per thread
#pragma unroll
  for (int it = 0; it < 4; it++) {
    int j = tid + it * 128;
    int r = j >> 3, c8 = j & 7;
    async_b128(qsBase + r * 144 + c8 * 16,
               qp + (long long)(qbase + r) * GPT_DH + c8 * 8);
  }
  wait_async0();
  __syncthreads();

  v8f Oacc[4] = {};
  float mrun[8], lrun[8];
#pragma unroll
  for (int s = 0; s < 8; s++) { mrun[s] = -INFINITY; lrun[s] = 0.f; }

  int nk = qbase + 64;  // causal extent for this block
  for (int kt = 0; kt < nk; kt += 32) {
    // K tile 32x64: async, 2 x b128 per thread
#pragma unroll
    for (int it = 0; it < 2; it++) {
      int j = tid + it * 128;
      int r = j >> 3, c8 = j & 7;
      async_b128(ksBase + r * 144 + c8 * 16,
                 kp + (long long)(kt + r) * GPT_DH + c8 * 8);
    }
    // V tile transposed to [d][key]: 8 dword pairs per thread
#pragma unroll
    for (int it = 0; it < 8; it++) {
      int j = tid + it * 128;          // 0..1023
      int r = j >> 5, cp = j & 31;     // key, d-pair
      unsigned int u = *(const unsigned int*)(vp + (long long)(kt + r) * GPT_DH + 2 * cp);
      Vs[2 * cp][r]     = (unsigned short)(u & 0xffffu);
      Vs[2 * cp + 1][r] = (unsigned short)(u >> 16);
    }
    wait_async0();
    __syncthreads();

    // S = Q(16x64) @ K^T(64x32): 2 key sub-tiles x 2 d-steps
    v8f S[2] = {};
#pragma unroll
    for (int j = 0; j < 2; j++) {
#pragma unroll
      for (int ds = 0; ds < 2; ds++) {
        Frag16 ua;
        int m = wv * 16 + lr;
#pragma unroll
        for (int vvi = 0; vvi < 8; vvi++) {
          int kb = (vvi < 4 ? 2 * vvi : 16 + 2 * (vvi - 4)) + hf * 8 + ds * 32;
          ua.u[vvi] = *(const unsigned int*)&Qs[m][kb];
        }
        Frag16 ub;
        int n = j * 16 + lr;
#pragma unroll
        for (int vvi = 0; vvi < 8; vvi++) {
          int kb = (vvi < 4 ? 2 * vvi : 16 + 2 * (vvi - 4)) + hf * 8 + ds * 32;
          ub.u[vvi] = *(const unsigned int*)&Ks[n][kb];
        }
        S[j] = __builtin_amdgcn_wmma_f32_16x16x32_bf16(
            false, ua.v, false, ub.v, (short)0, S[j], false, false);
      }
    }

    // online softmax per row (row = s + hf*8 within the wave's 16 rows)
    float alpha[8];
#pragma unroll
    for (int s = 0; s < 8; s++) {
      int row = qbase + wv * 16 + s + hf * 8;
      float s0 = S[0][s] * scale;
      float s1 = S[1][s] * scale;
      int c0 = kt + lr, c1 = kt + 16 + lr;
      if (c0 > row) s0 = -INFINITY;
      if (c1 > row) s1 = -INFINITY;
      float t = fmaxf(s0, s1);
      for (int off = 8; off >= 1; off >>= 1) t = fmaxf(t, __shfl_xor(t, off, 32));
      float mnew = fmaxf(mrun[s], t);
      float a  = __expf(mrun[s] - mnew);
      float p0 = __expf(s0 - mnew);
      float p1 = __expf(s1 - mnew);
      float rs = p0 + p1;
      for (int off = 8; off >= 1; off >>= 1) rs += __shfl_xor(rs, off, 32);
      lrun[s] = lrun[s] * a + rs;
      mrun[s] = mnew;
      alpha[s] = a;
      Ps[wv][s + hf * 8][lr]      = f2bf(p0);
      Ps[wv][s + hf * 8][16 + lr] = f2bf(p1);
    }

#pragma unroll
    for (int jc = 0; jc < 4; jc++)
#pragma unroll
      for (int s = 0; s < 8; s++) Oacc[jc][s] *= alpha[s];

    // O += P(16x32) @ V(32x64)
    Frag16 up;
#pragma unroll
    for (int vvi = 0; vvi < 8; vvi++) {
      int kb = (vvi < 4 ? 2 * vvi : 16 + 2 * (vvi - 4)) + hf * 8;
      up.u[vvi] = *(const unsigned int*)&Ps[wv][lr][kb];
    }
#pragma unroll
    for (int jc = 0; jc < 4; jc++) {
      Frag16 ub;
      int n = jc * 16 + lr;  // d index
#pragma unroll
      for (int vvi = 0; vvi < 8; vvi++) {
        int kb = (vvi < 4 ? 2 * vvi : 16 + 2 * (vvi - 4)) + hf * 8;  // key pair
        ub.u[vvi] = *(const unsigned int*)&Vs[n][kb];
      }
      Oacc[jc] = __builtin_amdgcn_wmma_f32_16x16x32_bf16(
          false, up.v, false, ub.v, (short)0, Oacc[jc], false, false);
    }
    __syncthreads();
  }

  // write normalized output (bf16), concat-head layout [B,T,H*64]
#pragma unroll
  for (int jc = 0; jc < 4; jc++) {
#pragma unroll
    for (int s = 0; s < 8; s++) {
      int row = qbase + wv * 16 + s + hf * 8;
      int col = h * GPT_DH + jc * 16 + lr;
      o[(long long)(b * GPT_T + row) * (GPT_H * GPT_DH) + col] =
          f2bf(Oacc[jc][s] / lrun[s]);
    }
  }
}

// ------------------------------------------------------------------
// Host-side orchestration
// ------------------------------------------------------------------
extern "C" void kernel_launch(void* const* d_in, const int* in_sizes, int n_in,
                              void* d_out, int out_size, void* d_ws, size_t ws_size,
                              hipStream_t stream) {
  (void)in_sizes; (void)n_in; (void)out_size; (void)ws_size;
  const int*   idx     = (const int*)  d_in[0];
  const float* tok_emb = (const float*)d_in[1];
  const float* pos_emb = (const float*)d_in[2];
  const float* ln1_s   = (const float*)d_in[3];
  const float* ln1_b   = (const float*)d_in[4];
  const float* wq      = (const float*)d_in[5];
  const float* wk      = (const float*)d_in[6];
  const float* wvp     = (const float*)d_in[7];
  const float* wo      = (const float*)d_in[8];
  const float* bo      = (const float*)d_in[9];
  const float* ln2_s   = (const float*)d_in[10];
  const float* ln2_b   = (const float*)d_in[11];
  const float* w1      = (const float*)d_in[12];
  const float* b1      = (const float*)d_in[13];
  const float* w2      = (const float*)d_in[14];
  const float* b2      = (const float*)d_in[15];
  const float* lnf_s   = (const float*)d_in[16];
  const float* lnf_b   = (const float*)d_in[17];
  const float* head_w  = (const float*)d_in[18];
  const float* head_b  = (const float*)d_in[19];
  float* out = (float*)d_out;

  // workspace: x fp32; activations bf16 (~35 MB total)
  const size_t NE = (size_t)GPT_M * GPT_C;  // 1.57M elements
  float* x = (float*)d_ws;
  unsigned short* hb = (unsigned short*)(x + NE);
  unsigned short* qb = hb + NE;
  unsigned short* kb = qb + NE;
  unsigned short* vb = kb + NE;
  unsigned short* ob = vb + NE;
  unsigned short* ff = ob + NE;   // [M, FF] bf16

  const float scale = 1.0f / sqrtf((float)GPT_C);  // reference scales by sqrt(C)

  embed_kernel<<<GPT_M, 256, 0, stream>>>(idx, tok_emb, pos_emb, x);

  for (int l = 0; l < GPT_L; l++) {
    ln_kernel<<<GPT_M, 256, 0, stream>>>(x, hb, ln1_s + l * GPT_C, ln1_b + l * GPT_C);

    // QKV: B is (H,C,DH); output bf16 mapped to [B,H,T,DH]
    const float* W3[3] = { wq + (size_t)l * GPT_H * GPT_C * GPT_DH,
                           wk + (size_t)l * GPT_H * GPT_C * GPT_DH,
                           wvp + (size_t)l * GPT_H * GPT_C * GPT_DH };
    unsigned short* O3[3] = { qb, kb, vb };
    for (int i = 0; i < 3; i++) {
      gemm_bf16<<<dim3(GPT_C / 128, GPT_M / 128), 256, 0, stream>>>(
          hb, W3[i], O3[i],
          GPT_M, GPT_C, GPT_C, GPT_C,
          /*sb_k*/ GPT_DH, /*sb_h*/ (long long)GPT_C * GPT_DH, /*sb_d*/ 1, /*dhB*/ GPT_DH,
          /*mrowC*/ GPT_T, /*sc_b*/ (long long)GPT_H * GPT_T * GPT_DH,
          /*sc_t*/ GPT_DH, /*dhC*/ GPT_DH, /*sc_h*/ (long long)GPT_T * GPT_DH, /*sc_d*/ 1,
          nullptr, nullptr, 0, 0, /*out_bf16*/ 1);
    }

    attn_kernel<<<dim3(GPT_T / 64, GPT_B * GPT_H), 128, 0, stream>>>(
        qb, kb, vb, ob, scale);

    // x = x + ob @ wo[l] + bo[l]   (fp32 residual stream)
    gemm_bf16<<<dim3(GPT_C / 128, GPT_M / 128), 256, 0, stream>>>(
        ob, wo + (size_t)l * GPT_C * GPT_C, x,
        GPT_M, GPT_C, GPT_C, GPT_C,
        GPT_C, 0, 1, GPT_C,
        GPT_M, 0, GPT_C, GPT_C, 0, 1,
        bo + l * GPT_C, x, GPT_C, 0, 0);

    ln_kernel<<<GPT_M, 256, 0, stream>>>(x, hb, ln2_s + l * GPT_C, ln2_b + l * GPT_C);

    // ff = relu(hb @ w1[l] + b1[l])  (bf16 out)
    gemm_bf16<<<dim3(GPT_FF / 128, GPT_M / 128), 256, 0, stream>>>(
        hb, w1 + (size_t)l * GPT_C * GPT_FF, ff,
        GPT_M, GPT_FF, GPT_C, GPT_C,
        GPT_FF, 0, 1, GPT_FF,
        GPT_M, 0, GPT_FF, GPT_FF, 0, 1,
        b1 + l * GPT_FF, nullptr, 0, 1, 1);

    // x = x + ff @ w2[l] + b2[l]
    gemm_bf16<<<dim3(GPT_C / 128, GPT_M / 128), 256, 0, stream>>>(
        ff, w2 + (size_t)l * GPT_FF * GPT_C, x,
        GPT_M, GPT_C, GPT_FF, GPT_FF,
        GPT_C, 0, 1, GPT_C,
        GPT_M, 0, GPT_C, GPT_C, 0, 1,
        b2 + l * GPT_C, x, GPT_C, 0, 0);
  }

  ln_kernel<<<GPT_M, 256, 0, stream>>>(x, hb, lnf_s, lnf_b);

  // logits = hb @ head_w + head_b   (dominant GEMM, fp32 out to d_out)
  gemm_bf16<<<dim3(GPT_V / 128, GPT_M / 128), 256, 0, stream>>>(
      hb, head_w, out,
      GPT_M, GPT_V, GPT_C, GPT_C,
      GPT_V, 0, 1, GPT_V,
      GPT_M, 0, GPT_V, GPT_V, 0, 1,
      head_b, nullptr, 0, 0, 0);
}
